// AttentiveNCF_81003083203132
// MI455X (gfx1250) — matrix operand order repaired and checked by reading, not compiled, not run on hardware
//
#include <hip/hip_runtime.h>
#include <hip/hip_bf16.h>

typedef __bf16 bf16;
typedef __attribute__((ext_vector_type(16))) __bf16 v16bf;
typedef __attribute__((ext_vector_type(8)))  __bf16 v8bf;
typedef __attribute__((ext_vector_type(8)))  float  v8f;
typedef __attribute__((ext_vector_type(4)))  float  v4f;

constexpr int Qn = 4096;
constexpr int Nn = 32768;
constexpr int Dn = 128;

// ---------------------------------------------------------------- helpers

static __device__ __forceinline__ v8f wmma_bf16(v16bf a, v16bf b, v8f c) {
  // (neg_a, A, neg_b, B, c_mod, C, reuse_a, reuse_b)
  return __builtin_amdgcn_wmma_f32_16x16x32_bf16(false, a, false, b, (short)0, c,
                                                 false, false);
}

static __device__ __forceinline__ v16bf cat8(v8bf lo, v8bf hi) {
  v16bf r;
#pragma unroll
  for (int i = 0; i < 8; i++) { r[i] = lo[i]; r[i + 8] = hi[i]; }
  return r;
}

// B operand (32x16 bf16): lane holds col n, K = half*16 + 0..15 contiguous.
static __device__ __forceinline__ v16bf load_b_bf16(const bf16* p) {
  return cat8(*(const v8bf*)p, *(const v8bf*)(p + 8));
}

// B operand sourced from f32 row-major B^T: 16 contiguous floats -> bf16.
static __device__ __forceinline__ v16bf cvt_b_f32(const float* p) {
  v4f x0 = *(const v4f*)(p + 0), x1 = *(const v4f*)(p + 4);
  v4f x2 = *(const v4f*)(p + 8), x3 = *(const v4f*)(p + 12);
  v16bf r;
#pragma unroll
  for (int i = 0; i < 4; i++) {
    r[i]      = (bf16)x0[i];
    r[i + 4]  = (bf16)x1[i];
    r[i + 8]  = (bf16)x2[i];
    r[i + 12] = (bf16)x3[i];
  }
  return r;
}

// A operand (16x32 bf16): lane holds row m; elems 0..15 = K half*16 + 0..15 contiguous (same as B)
static __device__ __forceinline__ v16bf load_a_bf16(const bf16* p0, const bf16* p1) {
  return cat8(*(const v8bf*)p0, *(const v8bf*)p1);
}

static __device__ __forceinline__ v16bf cvt_a_f32(const float* p0, const float* p1) {
  v4f x0 = *(const v4f*)(p0 + 0), x1 = *(const v4f*)(p0 + 4);
  v4f y0 = *(const v4f*)(p1 + 0), y1 = *(const v4f*)(p1 + 4);
  v16bf r;
#pragma unroll
  for (int i = 0; i < 4; i++) {
    r[i]      = (bf16)x0[i];
    r[i + 4]  = (bf16)x1[i];
    r[i + 8]  = (bf16)y0[i];
    r[i + 12] = (bf16)y1[i];
  }
  return r;
}

// ---------------------------------------------------------------- kernel 1: e1 -> bf16

__global__ void k_cvt_e1(const float* __restrict__ e1, bf16* __restrict__ e1b) {
  int i = blockIdx.x * 256 + threadIdx.x;
  e1b[i] = (bf16)e1[i];
}

// ------------------------------------------- kernel 2: e2 -> e2^T (bf16) + partial sums

__global__ void __launch_bounds__(256) k_transpose(const float* __restrict__ e2,
                                                   bf16* __restrict__ e2t,
                                                   float* __restrict__ partials) {
  __shared__ float tile[64][129];  // +1 pad: conflict-free column reads
  int tid = threadIdx.x;
  int n0 = blockIdx.x * 64;
#pragma unroll
  for (int j = 0; j < 32; j++) {
    int i = tid + j * 256;
    int r = i >> 7, c = i & 127;
    tile[r][c] = e2[(size_t)(n0 + r) * Dn + c];
  }
  __syncthreads();
#pragma unroll
  for (int j = 0; j < 32; j++) {
    int i = tid + j * 256;
    int d = i >> 6, rr = i & 63;
    e2t[(size_t)d * Nn + n0 + rr] = (bf16)tile[rr][d];
  }
  if (tid < 128) {
    float s = 0.f;
#pragma unroll
    for (int r = 0; r < 64; r++) s += tile[r][tid];
    partials[blockIdx.x * Dn + tid] = s;  // deterministic: no atomics
  }
}

// --------------------------------- kernel 3: finish sum_e2, svec1 = W1 * sum_e2

__global__ void k_reduce(const float* __restrict__ partials,
                         const float* __restrict__ W1,
                         float* __restrict__ sum_e2, float* __restrict__ svec1) {
  __shared__ float s[128];
  int d = threadIdx.x;
  float acc = 0.f;
  for (int b = 0; b < Nn / 64; b++) acc += partials[b * Dn + d];
  s[d] = acc;
  sum_e2[d] = acc;
  __syncthreads();
  float v = 0.f;
#pragma unroll 8
  for (int k = 0; k < Dn; k++) v += s[k] * W1[(size_t)d * Dn + k];
  svec1[d] = v;
}

// ------------------------- kernel 4: hidden = relu(e2 @ attn_W^T + b) in bf16 (WMMA)

__global__ void __launch_bounds__(256) k_hidden(const float* __restrict__ e2,
                                                const float* __restrict__ attn_W,
                                                const float* __restrict__ attn_b,
                                                bf16* __restrict__ hid) {
  int tid = threadIdx.x;
  int lane = tid & 31;
  int w = tid >> 5;           // wave 0..7, 16 rows each -> 128 rows per block
  int half = lane >> 4;
  int c = lane & 15;
  int rowbase = blockIdx.x * 128 + w * 16;
  const float* arow = e2 + (size_t)(rowbase + c) * Dn;

  v16bf a[4];
#pragma unroll
  for (int kc = 0; kc < 4; kc++)
    a[kc] = cvt_a_f32(arow + kc * 32 + half * 16, arow + kc * 32 + half * 16 + 8);

#pragma unroll
  for (int dt = 0; dt < 8; dt++) {
    int dcol = dt * 16 + c;
    v8f acc = {};
#pragma unroll
    for (int kc = 0; kc < 4; kc++) {
      v16bf b = cvt_b_f32(attn_W + (size_t)dcol * Dn + kc * 32 + half * 16);
      acc = wmma_bf16(a[kc], b, acc);
    }
    float bias = attn_b[dcol];
    bf16* outp = hid + (size_t)rowbase * Dn + dcol;
#pragma unroll
    for (int v = 0; v < 8; v++) {
      float x = fmaxf(acc[v] + bias, 0.f);
      outp[(size_t)(v + 8 * half) * Dn] = (bf16)x;
    }
  }
}

// --------------------- kernel 5: fused flash attention + message aggregation

__global__ void __launch_bounds__(128) k_flash(
    const float* __restrict__ e1, const bf16* __restrict__ e1b,
    const bf16* __restrict__ hid, const bf16* __restrict__ e2t,
    const float* __restrict__ W1, const float* __restrict__ W2,
    const float* __restrict__ sum_e2, const float* __restrict__ svec1,
    float* __restrict__ out) {
  __shared__ bf16 buf[4][16][128];  // per-wave staging (P transpose / epilogue A)
  int tid = threadIdx.x;
  int lane = tid & 31;
  int w = tid >> 5;  // 0..3 waves, 16 query rows each
  int half = lane >> 4;
  int c = lane & 15;
  int qb = blockIdx.x * 64 + w * 16;

  // Preload query tile as A operands (K = D = 128 -> 4 chunks of 32)
  const bf16* qrow = e1b + (size_t)(qb + c) * Dn;
  v16bf aq[4];
#pragma unroll
  for (int kc = 0; kc < 4; kc++)
    aq[kc] = load_a_bf16(qrow + kc * 32 + half * 16, qrow + kc * 32 + half * 16 + 8);

  v8f acc[8];
  v8f zero = {};
#pragma unroll
  for (int dt = 0; dt < 8; dt++) acc[dt] = zero;
  float m[8], l[8];
#pragma unroll
  for (int v = 0; v < 8; v++) { m[v] = -3.0e38f; l[v] = 0.f; }

#pragma unroll 1
  for (int nb = 0; nb < Nn; nb += 32) {
    // S = e1_tile @ hidden_chunk^T : two 16x16 C tiles
    v8f s0 = zero, s1 = zero;
    const bf16* hrow0 = hid + (size_t)(nb + c) * Dn;
    const bf16* hrow1 = hid + (size_t)(nb + 16 + c) * Dn;
#pragma unroll
    for (int kc = 0; kc < 4; kc++)
      s0 = wmma_bf16(aq[kc], load_b_bf16(hrow0 + kc * 32 + half * 16), s0);
#pragma unroll
    for (int kc = 0; kc < 4; kc++)
      s1 = wmma_bf16(aq[kc], load_b_bf16(hrow1 + kc * 32 + half * 16), s1);

    // Online softmax (rows live across the 16-lane half; xor masks 1..8 stay in-half)
#pragma unroll
    for (int v = 0; v < 8; v++) {
      float x = fmaxf(s0[v], s1[v]);
#pragma unroll
      for (int off = 8; off > 0; off >>= 1) x = fmaxf(x, __shfl_xor(x, off, 32));
      float mn = fmaxf(m[v], x);
      float scale = __expf(m[v] - mn);
      float p0 = __expf(s0[v] - mn);
      float p1 = __expf(s1[v] - mn);
      float rs = p0 + p1;
#pragma unroll
      for (int off = 8; off > 0; off >>= 1) rs += __shfl_xor(rs, off, 32);
      l[v] = l[v] * scale + rs;
      m[v] = mn;
#pragma unroll
      for (int dt = 0; dt < 8; dt++) acc[dt][v] *= scale;
      int r = v + 8 * half;
      buf[w][r][c] = (bf16)p0;       // C layout -> row-major P in LDS
      buf[w][r][16 + c] = (bf16)p1;
    }
    __syncthreads();
    v16bf pa = load_a_bf16(&buf[w][c][half * 16], &buf[w][c][half * 16 + 8]);
    __syncthreads();

    // O += P @ V : B operand from e2^T rows (contiguous over neighbor axis)
#pragma unroll
    for (int dt = 0; dt < 8; dt++) {
      const bf16* vrow = e2t + (size_t)(dt * 16 + c) * Nn + nb + half * 16;
      acc[dt] = wmma_bf16(pa, load_b_bf16(vrow), acc[dt]);
    }
  }

  // ---- epilogue: a = e1 + O/l ; out = a@W1^T + svec1 + (a*sum_e2)@W2^T ; leaky
  v8f o[8];
#pragma unroll
  for (int dt = 0; dt < 8; dt++) o[dt] = zero;
  float se_col[8];

#pragma unroll
  for (int dt = 0; dt < 8; dt++) {
    int dcol = dt * 16 + c;
    se_col[dt] = sum_e2[dcol];
#pragma unroll
    for (int v = 0; v < 8; v++) {
      int r = v + 8 * half;
      float aval = e1[(size_t)(qb + r) * Dn + dcol] + acc[dt][v] / l[v];
      acc[dt][v] = aval;  // keep f32 copy for phase 2
      buf[w][r][dcol] = (bf16)aval;
    }
  }
  __syncthreads();
  {
    const bf16* arow = &buf[w][c][0];
#pragma unroll
    for (int kc = 0; kc < 4; kc++) {
      v16bf aa = load_a_bf16(arow + kc * 32 + half * 16, arow + kc * 32 + half * 16 + 8);
#pragma unroll
      for (int dt = 0; dt < 8; dt++) {
        v16bf bw = cvt_b_f32(W1 + (size_t)(dt * 16 + c) * Dn + kc * 32 + half * 16);
        o[dt] = wmma_bf16(aa, bw, o[dt]);
      }
    }
  }
  __syncthreads();
#pragma unroll
  for (int dt = 0; dt < 8; dt++)
#pragma unroll
    for (int v = 0; v < 8; v++)
      buf[w][v + 8 * half][dt * 16 + c] = (bf16)(acc[dt][v] * se_col[dt]);
  __syncthreads();
  {
    const bf16* arow = &buf[w][c][0];
#pragma unroll
    for (int kc = 0; kc < 4; kc++) {
      v16bf aa = load_a_bf16(arow + kc * 32 + half * 16, arow + kc * 32 + half * 16 + 8);
#pragma unroll
      for (int dt = 0; dt < 8; dt++) {
        v16bf bw = cvt_b_f32(W2 + (size_t)(dt * 16 + c) * Dn + kc * 32 + half * 16);
        o[dt] = wmma_bf16(aa, bw, o[dt]);
      }
    }
  }
#pragma unroll
  for (int dt = 0; dt < 8; dt++) {
    int dcol = dt * 16 + c;
    float sv = svec1[dcol];
#pragma unroll
    for (int v = 0; v < 8; v++) {
      float x = o[dt][v] + sv;
      x = (x > 0.f) ? x : 0.01f * x;
      out[(size_t)(qb + v + 8 * half) * Dn + dcol] = x;
    }
  }
}

// ---------------------------------------------------------------- launcher

extern "C" void kernel_launch(void* const* d_in, const int* in_sizes, int n_in,
                              void* d_out, int out_size, void* d_ws, size_t ws_size,
                              hipStream_t stream) {
  const float* e1     = (const float*)d_in[0];
  const float* e2     = (const float*)d_in[1];
  const float* attn_W = (const float*)d_in[2];
  const float* attn_b = (const float*)d_in[3];
  const float* W1     = (const float*)d_in[4];
  const float* W2     = (const float*)d_in[5];
  float* out = (float*)d_out;

  char* ws = (char*)d_ws;
  // workspace layout (bytes), total ~17.3 MB
  bf16*  e1b      = (bf16*)(ws + 0);          // 4096*128*2   = 1 MB
  bf16*  hid      = (bf16*)(ws + 1048576u);   // 32768*128*2  = 8 MB
  bf16*  e2t      = (bf16*)(ws + 9437184u);   // 128*32768*2  = 8 MB
  float* partials = (float*)(ws + 17825792u); // 512*128*4    = 256 KB
  float* sum_e2   = (float*)(ws + 18087936u); // 128*4
  float* svec1    = (float*)(ws + 18088448u); // 128*4

  k_cvt_e1<<<(Qn * Dn) / 256, 256, 0, stream>>>(e1, e1b);
  k_transpose<<<Nn / 64, 256, 0, stream>>>(e2, e2t, partials);
  k_reduce<<<1, 128, 0, stream>>>(partials, W1, sum_e2, svec1);
  k_hidden<<<Nn / 128, 256, 0, stream>>>(e2, attn_W, attn_b, hid);
  k_flash<<<Qn / 64, 128, 0, stream>>>(e1, e1b, hid, e2t, W1, W2, sum_e2, svec1, out);
}
